// TTLoRALinearWrapperContraction_86749749444633
// MI455X (gfx1250) — matrix operand order repaired and checked by reading, not compiled, not run on hardware
//
#include <hip/hip_runtime.h>

typedef float    v2f  __attribute__((ext_vector_type(2)));
typedef float    v4f  __attribute__((ext_vector_type(4)));
typedef float    v8f  __attribute__((ext_vector_type(8)));
typedef _Float16 v4h  __attribute__((ext_vector_type(4)));
typedef _Float16 v8h  __attribute__((ext_vector_type(8)));
typedef _Float16 v16h __attribute__((ext_vector_type(16)));

#define TT_ALPHA 16.0f
#define D_DIM 2048
#define M_TOTAL 16384   // B*S = 4*4096
#define BK 32           // K-chunk staged in LDS (one 16x16x32 WMMA K-step)

// ---------------------------------------------------------------------------
// Kernel 1: TT update. One block per token. Writes ALPHA * update to d_out.
// (rank-8 contractions; 1.6% of total FLOPs -> VALU fp32, exact)
// ---------------------------------------------------------------------------
__global__ __launch_bounds__(256) void tt_update_kernel(
    const float* __restrict__ x,
    const float* __restrict__ c0, const float* __restrict__ c1,
    const float* __restrict__ c2, const float* __restrict__ c3,
    const float* __restrict__ c4, const float* __restrict__ c5,
    float* __restrict__ out)
{
    __shared__ __align__(16) float xs[2048];
    __shared__ __align__(16) float t1[2048];
    __shared__ float t2[128];
    __shared__ float t3[8];
    __shared__ float s0[64];
    __shared__ float s1[1024];
    __shared__ float s2[1024];
    __shared__ float s3[1024];
    __shared__ float s4[1024];
    __shared__ float s5[64];

    const int tid = threadIdx.x;
    const long token = blockIdx.x;
    const float* xt = x + token * D_DIM;

    for (int i = tid; i < D_DIM / 4; i += 256)
        ((v4f*)xs)[i] = ((const v4f*)xt)[i];
    if (tid < 64)  s0[tid] = c0[tid];
    for (int i = tid; i < 1024; i += 256) {
        s1[i] = c1[i]; s2[i] = c2[i]; s3[i] = c3[i]; s4[i] = c4[i];
    }
    if (tid >= 64 && tid < 128) s5[tid - 64] = c5[tid - 64];
    __syncthreads();

    // ---- input contraction 1: thread = (i0,i1), tid = i0*16+i1 ----
    {
        const float* xv = &xs[tid * 8];
        float acc[8];
#pragma unroll
        for (int p = 0; p < 8; ++p) acc[p] = 0.0f;
#pragma unroll
        for (int i2 = 0; i2 < 8; ++i2) {
            const float xvv = xv[i2];
#pragma unroll
            for (int p = 0; p < 8; ++p) acc[p] += xvv * s0[i2 * 8 + p];
        }
#pragma unroll
        for (int p = 0; p < 8; ++p) t1[p * 256 + tid] = acc[p];
    }
    __syncthreads();

    // ---- input contraction 2: 128 outputs ----
    if (tid < 128) {
        const int i0 = tid >> 3, p = tid & 7;
        float acc = 0.0f;
        for (int r = 0; r < 8; ++r) {
#pragma unroll
            for (int i1 = 0; i1 < 16; ++i1)
                acc += t1[r * 256 + i0 * 16 + i1] * s1[r * 128 + i1 * 8 + p];
        }
        t2[p * 16 + i0] = acc;
    }
    __syncthreads();

    // ---- input contraction 3: 8 outputs ----
    if (tid < 8) {
        const int p = tid;
        float acc = 0.0f;
        for (int r = 0; r < 8; ++r) {
#pragma unroll
            for (int i0 = 0; i0 < 16; ++i0)
                acc += t2[r * 16 + i0] * s2[r * 128 + i0 * 8 + p];
        }
        t3[p] = acc;
    }
    __syncthreads();

    // ---- output expansion: thread owns (n0,n1), writes 8 contiguous floats ----
    {
        const int n0 = tid >> 4, n1 = tid & 15;
        float t4[8], t5[8];
#pragma unroll
        for (int q = 0; q < 8; ++q) t4[q] = 0.0f;
#pragma unroll
        for (int r = 0; r < 8; ++r) {
            const float v = t3[r];
#pragma unroll
            for (int q = 0; q < 8; ++q) t4[q] += v * s3[r * 128 + n0 * 8 + q];
        }
#pragma unroll
        for (int q = 0; q < 8; ++q) t5[q] = 0.0f;
#pragma unroll
        for (int r = 0; r < 8; ++r) {
            const float v = t4[r];
#pragma unroll
            for (int q = 0; q < 8; ++q) t5[q] += v * s4[r * 128 + n1 * 8 + q];
        }
        float res[8];
#pragma unroll
        for (int n2 = 0; n2 < 8; ++n2) {
            float acc = 0.0f;
#pragma unroll
            for (int r = 0; r < 8; ++r) acc += t5[r] * s5[r * 8 + n2];
            res[n2] = acc * TT_ALPHA;
        }
        float* op = out + token * D_DIM + n0 * 128 + n1 * 8;
        v4f r0 = { res[0], res[1], res[2], res[3] };
        v4f r1 = { res[4], res[5], res[6], res[7] };
        ((v4f*)op)[0] = r0;
        ((v4f*)op)[1] = r1;
    }
}

// ---------------------------------------------------------------------------
// Kernel 2: C = X @ W^T + bias + C  via f16x3 error-corrected split on
// V_WMMA_F32_16X16X32_F16:  X=Xh+Xl, W=Wh+Wl,  X*W ~= Xh*Wh + Xh*Wl + Xl*Wh
// (dropped Xl*Wl term ~2^-22 relative; fp32 accumulate -> fp32-class accuracy)
//
// Block tile 128x128, 8 waves as 2(m) x 4(n); wave tile 64x32 = 4x2 accums.
// ISA 16-bit fragment layouts (CDNA5 ISA, wave32):
//   A 16x32: lane l -> M=l%16, h=l/16; elems 0..7 <-> K=8h..8h+7,
//            elems 8..15 <-> K=16+8h..16+8h+7          (2x ds_load_b128)
//   B 32x16: lane l -> N=l%16, h=l/16; same K mapping as A (2x ds_load_b128)
//   C 16x16 f32: VGPR v -> M = v + 8h, N = l%16
// ---------------------------------------------------------------------------
__global__ __launch_bounds__(256) void gemm_wmma_f16x3_kernel(
    const float* __restrict__ X, const float* __restrict__ W,
    const float* __restrict__ bias, float* __restrict__ out)
{
    __shared__ __align__(16) _Float16 Xh[128 * BK];  // [m][k] hi plane
    __shared__ __align__(16) _Float16 Xl[128 * BK];  // [m][k] lo plane
    __shared__ __align__(16) _Float16 Wh[128 * BK];  // [n][k] hi plane
    __shared__ __align__(16) _Float16 Wl[128 * BK];  // [n][k] lo plane

    const int tid  = threadIdx.x;
    const int lane = tid & 31;
    const int wave = tid >> 5;
    const int wm   = wave >> 2;    // 0..1  (64-row slab)
    const int wn   = wave & 3;     // 0..3  (32-col slab)
    const int half = lane >> 4;
    const int l16  = lane & 15;
    const long m0 = (long)blockIdx.x * 128;
    const long n0 = (long)blockIdx.y * 128;

    v8f acc[4][2];
#pragma unroll
    for (int mt = 0; mt < 4; ++mt)
#pragma unroll
        for (int nt = 0; nt < 2; ++nt)
            acc[mt][nt] = {};

    for (int k0 = 0; k0 < D_DIM; k0 += BK) {
        // Stage + split-convert one K-chunk: 4096 floats of X and of W.
#pragma unroll
        for (int i = 0; i < 4; ++i) {
            const int c   = tid + i * 256;  // 0..1023 float4 chunks
            const int row = c >> 3;
            const int kq  = (c & 7) * 4;
            const v4f xv = *(const v4f*)&X[(m0 + row) * D_DIM + k0 + kq];
            const v4f wv = *(const v4f*)&W[(n0 + row) * D_DIM + k0 + kq];
            v4h xh, xl, wh, wl;
#pragma unroll
            for (int e = 0; e < 4; ++e) {
                xh[e] = (_Float16)xv[e];
                xl[e] = (_Float16)(xv[e] - (float)xh[e]);
                wh[e] = (_Float16)wv[e];
                wl[e] = (_Float16)(wv[e] - (float)wh[e]);
            }
            *(v4h*)&Xh[row * BK + kq] = xh;
            *(v4h*)&Xl[row * BK + kq] = xl;
            *(v4h*)&Wh[row * BK + kq] = wh;
            *(v4h*)&Wl[row * BK + kq] = wl;
        }
        if (k0 + BK < D_DIM) {  // warm next chunk into cache (global_prefetch_b8)
            __builtin_prefetch(&X[(m0 + (tid >> 1)) * D_DIM + k0 + BK], 0, 1);
            __builtin_prefetch(&W[(n0 + (tid >> 1)) * D_DIM + k0 + BK], 0, 1);
        }
        __syncthreads();

        // Fragments for this K=32 chunk.
        v16h ah[4], al[4], bh[2], bl[2];
#pragma unroll
        for (int mt = 0; mt < 4; ++mt) {
            const int row = wm * 64 + mt * 16 + l16;
            const v8h h0 = *(const v8h*)&Xh[row * BK + half * 8];
            const v8h h1 = *(const v8h*)&Xh[row * BK + 16 + half * 8];
            const v8h l0 = *(const v8h*)&Xl[row * BK + half * 8];
            const v8h l1 = *(const v8h*)&Xl[row * BK + 16 + half * 8];
            ah[mt] = __builtin_shufflevector(h0, h1, 0,1,2,3,4,5,6,7,8,9,10,11,12,13,14,15);
            al[mt] = __builtin_shufflevector(l0, l1, 0,1,2,3,4,5,6,7,8,9,10,11,12,13,14,15);
        }
#pragma unroll
        for (int nt = 0; nt < 2; ++nt) {
            const int col = wn * 32 + nt * 16 + l16;
            const v8h h0 = *(const v8h*)&Wh[col * BK + half * 8];
            const v8h h1 = *(const v8h*)&Wh[col * BK + 16 + half * 8];
            const v8h l0 = *(const v8h*)&Wl[col * BK + half * 8];
            const v8h l1 = *(const v8h*)&Wl[col * BK + 16 + half * 8];
            bh[nt] = __builtin_shufflevector(h0, h1, 0,1,2,3,4,5,6,7,8,9,10,11,12,13,14,15);
            bl[nt] = __builtin_shufflevector(l0, l1, 0,1,2,3,4,5,6,7,8,9,10,11,12,13,14,15);
        }

        // 3-term split accumulation, all in fp32 C.
#pragma unroll
        for (int mt = 0; mt < 4; ++mt)
#pragma unroll
            for (int nt = 0; nt < 2; ++nt) {
                acc[mt][nt] = __builtin_amdgcn_wmma_f32_16x16x32_f16(
                    false, ah[mt], false, bl[nt], (short)0, acc[mt][nt], false, false);
                acc[mt][nt] = __builtin_amdgcn_wmma_f32_16x16x32_f16(
                    false, al[mt], false, bh[nt], (short)0, acc[mt][nt], false, false);
                acc[mt][nt] = __builtin_amdgcn_wmma_f32_16x16x32_f16(
                    false, ah[mt], false, bh[nt], (short)0, acc[mt][nt], false, false);
            }
        __syncthreads();
    }

    // Epilogue: out = gemm + bias + out (out already holds ALPHA*update).
#pragma unroll
    for (int mt = 0; mt < 4; ++mt) {
#pragma unroll
        for (int nt = 0; nt < 2; ++nt) {
            const long n = n0 + wn * 32 + nt * 16 + l16;
            const float bv = bias[n];
#pragma unroll
            for (int v = 0; v < 8; ++v) {
                const long m = m0 + wm * 64 + mt * 16 + v + 8 * half;
                const long idx = m * D_DIM + n;
                out[idx] = acc[mt][nt][v] + bv + out[idx];
            }
        }
    }
}

// ---------------------------------------------------------------------------
extern "C" void kernel_launch(void* const* d_in, const int* in_sizes, int n_in,
                              void* d_out, int out_size, void* d_ws, size_t ws_size,
                              hipStream_t stream) {
    const float* x  = (const float*)d_in[0];
    const float* W  = (const float*)d_in[1];
    const float* b  = (const float*)d_in[2];

    const float *c0, *c1, *c2, *c3, *c4, *c5;
    if (n_in >= 9) {
        c0 = (const float*)d_in[3]; c1 = (const float*)d_in[4];
        c2 = (const float*)d_in[5]; c3 = (const float*)d_in[6];
        c4 = (const float*)d_in[7]; c5 = (const float*)d_in[8];
    } else {
        // cores delivered as one flat buffer: sizes 64,1024,1024,1024,1024,64
        const float* cz = (const float*)d_in[3];
        c0 = cz;        c1 = cz + 64;   c2 = cz + 1088;
        c3 = cz + 2112; c4 = cz + 3136; c5 = cz + 4160;
    }
    float* out = (float*)d_out;

    // Pass 1: out = ALPHA * TT_update(x)   (fully overwrites d_out)
    tt_update_kernel<<<M_TOTAL, 256, 0, stream>>>(x, c0, c1, c2, c3, c4, c5, out);

    // Pass 2: out += x @ W^T + bias   (f16x3 split WMMA)
    dim3 grid(M_TOTAL / 128, D_DIM / 128);
    gemm_wmma_f16x3_kernel<<<grid, 256, 0, stream>>>(x, W, b, out);
}
